// SplitQKV_CIM_Wrapper_16329465659979
// MI455X (gfx1250) — matrix operand order repaired and checked by reading, not compiled, not run on hardware
//
#include <hip/hip_runtime.h>
#include <hip/hip_bf16.h>
#include <hip/hip_fp16.h>

// CDNA5 (gfx1250) CIM QKV projection.
// Fused single GEMM: [8192 x 768] (f16, int-valued quantized acts) x
// [768 x 2304] (f16, noised quantized weights, q|k|v concatenated) with
// per-64-row-group ADC quantization of f32 partial sums between WMMA pairs.
// Tiles are staged with GLOBAL_LOAD_ASYNC_TO_LDS (ASYNCcnt) — the gfx1250
// async copy path — instead of global->VGPR->ds_store.

typedef __attribute__((ext_vector_type(16))) _Float16 v16h;
typedef __attribute__((ext_vector_type(8)))  float    v8f;
typedef __attribute__((ext_vector_type(4)))  float    v4f;
typedef __attribute__((ext_vector_type(4)))  _Float16 v4h;

#define BT     8192   // B*T rows
#define CIN    768
#define NOUT   2304   // 3*768 concatenated
#define COUT1  768
#define GROUPS 12     // CIN / 64
#define GSIZE  64     // ROWS_PARALLEL
#define BM     128
#define BN     64
#define LDA    72     // padded LDS stride (halves); 144B row stride, 16B aligned
#define LDB    72

union F16Frag { uint4 u[2]; v16h h; };

// ---- gfx1250 async global->LDS copy (VGLOBAL, tracked by ASYNCcnt) ----
__device__ __forceinline__ void async_ld_b128(unsigned lds_byte_addr,
                                              const void* gbase,
                                              int byte_off) {
  // GVS mode: mem = SGPR64 + VGPR_I32; vdst VGPR holds LDS byte address.
  asm volatile("global_load_async_to_lds_b128 %0, %1, %2"
               :
               : "v"(lds_byte_addr), "v"(byte_off), "s"(gbase)
               : "memory");
}
__device__ __forceinline__ void wait_async0() {
  asm volatile("s_wait_asynccnt 0" ::: "memory");
}

// ---------------- scale reduction ----------------

__global__ void k_zero_scales(unsigned* s) {
  if (threadIdx.x < 4) s[threadIdx.x] = 0u;
}

// abs-max via monotone bit pattern of non-negative floats (deterministic).
__global__ void k_absmax(const float* __restrict__ src, int n,
                         unsigned* __restrict__ dst) {
  __shared__ float red[256];
  float m = 0.f;
  for (int i = blockIdx.x * blockDim.x + threadIdx.x; i < n;
       i += gridDim.x * blockDim.x)
    m = fmaxf(m, fabsf(src[i]));
  red[threadIdx.x] = m;
  __syncthreads();
  for (int s = 128; s > 0; s >>= 1) {
    if ((int)threadIdx.x < s)
      red[threadIdx.x] = fmaxf(red[threadIdx.x], red[threadIdx.x + s]);
    __syncthreads();
  }
  if (threadIdx.x == 0) atomicMax(dst, __float_as_uint(red[0]));
}

// ---------------- quantization prepass ----------------

__device__ __forceinline__ float qclip(float v) {
  // round-to-nearest-even (matches numpy/jax round) after clip to [-128,127]
  return rintf(fminf(fmaxf(v, -128.f), 127.f));
}

__global__ void k_quant_x(const float* __restrict__ x,
                          const unsigned* __restrict__ scales,
                          _Float16* __restrict__ xq) {
  const float sx  = __uint_as_float(scales[0]) * (1.f / 127.f);
  const float inv = 1.f / sx;
  size_t i = ((size_t)blockIdx.x * blockDim.x + threadIdx.x) * 4;
  if (i >= (size_t)BT * CIN) return;
  v4f v = *(const v4f*)(x + i);
  v4h o;
  o.x = (_Float16)qclip(v.x * inv);
  o.y = (_Float16)qclip(v.y * inv);
  o.z = (_Float16)qclip(v.z * inv);
  o.w = (_Float16)qclip(v.w * inv);
  *(v4h*)(xq + i) = o;
}

__global__ void k_quant_w(const float* __restrict__ w0, const float* __restrict__ w1,
                          const float* __restrict__ w2, const float* __restrict__ n0,
                          const float* __restrict__ n1, const float* __restrict__ n2,
                          const unsigned* __restrict__ scales,
                          _Float16* __restrict__ wqc) {
  const int m = blockIdx.y;
  const float* w  = (m == 0) ? w0 : (m == 1) ? w1 : w2;
  const float* nz = (m == 0) ? n0 : (m == 1) ? n1 : n2;
  const float sw  = __uint_as_float(scales[1 + m]) * (1.f / 127.f);
  const float inv = 1.f / sw;
  size_t i = ((size_t)blockIdx.x * blockDim.x + threadIdx.x) * 4;
  if (i >= (size_t)COUT1 * CIN) return;
  v4f v = *(const v4f*)(w + i);
  v4f z = *(const v4f*)(nz + i);
  v4h o;
  o.x = (_Float16)(qclip(v.x * inv) * (1.f + 0.05f * z.x));
  o.y = (_Float16)(qclip(v.y * inv) * (1.f + 0.05f * z.y));
  o.z = (_Float16)(qclip(v.z * inv) * (1.f + 0.05f * z.z));
  o.w = (_Float16)(qclip(v.w * inv) * (1.f + 0.05f * z.w));
  *(v4h*)(wqc + (size_t)m * COUT1 * CIN + i) = o;
}

// ---------------- WMMA GEMM with per-group ADC quantization ----------------

__global__ __launch_bounds__(256) void k_cim_gemm(
    const _Float16* __restrict__ xq, const _Float16* __restrict__ wqc,
    const unsigned* __restrict__ scales,
    const float* __restrict__ bq, const float* __restrict__ bk,
    const float* __restrict__ bv, float* __restrict__ out) {
  __shared__ _Float16 As[BM * LDA];  // [row][k], padded
  __shared__ _Float16 Bs[BN * LDB];  // [n][k],  padded (weights are N-major)

  const int tid  = threadIdx.x;
  const int wid  = tid >> 5;
  const int lane = tid & 31;
  const int nblk = blockIdx.x;  // 0..35
  const int mblk = blockIdx.y;  // 0..63

  const int wm = (wid & 3) * 32;   // wave row offset within macro tile
  const int wn = (wid >> 2) * 32;  // wave col offset within macro tile

  const int lrow  = lane & 15;
  const int hisel = lane >> 4;  // 0: lanes 0-15, 1: lanes 16-31

  const _Float16* gA = xq  + (size_t)(mblk * BM) * CIN;
  const _Float16* gB = wqc + (size_t)(nblk * BN) * CIN;

  // low 32 bits of a generic pointer into LDS == wave-relative LDS byte addr
  const unsigned ldsA = (unsigned)(uintptr_t)&As[0];
  const unsigned ldsB = (unsigned)(uintptr_t)&Bs[0];

  const float s_ps    = 64.f * 127.f;  // ADC LSB = ROWS_PARALLEL*qi*qw/adc_max
  const float inv_sps = 1.f / s_ps;

  // per-thread tile-copy coordinates (b128 granules)
  const int arow[4] = {tid >> 3, (tid + 256) >> 3, (tid + 512) >> 3, (tid + 768) >> 3};
  const int kc16 = (tid & 7) * 16;  // byte offset of the 16B granule within a 64-elem row

  v8f tot[2][2] = {{{}, {}}, {{}, {}}};

  for (int g = 0; g < GROUPS; ++g) {
    const int gbyte = g * GSIZE * 2;  // byte offset of this K-group
    __syncthreads();  // previous group's LDS reads complete before overwrite

    // ---- async stage: global -> LDS directly (no VGPR staging) ----
#pragma unroll
    for (int i = 0; i < 4; ++i) {  // 128x64 A tile: 1024 b128 granules
      int row = arow[i];
      async_ld_b128(ldsA + row * (LDA * 2) + kc16, gA,
                    gbyte + row * (CIN * 2) + kc16);
    }
#pragma unroll
    for (int i = 0; i < 2; ++i) {  // 64x64 B tile: 512 b128 granules
      int row = arow[i];
      async_ld_b128(ldsB + row * (LDB * 2) + kc16, gB,
                    gbyte + row * (CIN * 2) + kc16);
    }
    if (g + 1 < GROUPS) {  // warm caches for next K-group (global_prefetch_b8)
      __builtin_prefetch(gA + (g + 1) * GSIZE + (size_t)arow[0] * CIN, 0, 0);
      __builtin_prefetch(gB + (g + 1) * GSIZE + (size_t)arow[0] * CIN, 0, 0);
    }
    wait_async0();     // this wave's async transfers landed in LDS
    __syncthreads();   // all waves' transfers visible

    // ---- group partial sums: 2 WMMA K-steps of 32, f32 accumulate ----
    v8f acc[2][2] = {{{}, {}}, {{}, {}}};
#pragma unroll
    for (int ks = 0; ks < 2; ++ks) {
      F16Frag af[2], bf[2];
#pragma unroll
      for (int mi = 0; mi < 2; ++mi) {
        // A 16x32 f16 layout: lanes 0-15 rows M=0..15 hold K 0-7 & 16-23,
        // lanes 16-31 hold K 8-15 & 24-31 (ISA 7.12.2)
        int row = wm + mi * 16 + lrow;
        int kb  = ks * 32 + hisel * 8;
        af[mi].u[0] = *(const uint4*)&As[row * LDA + kb];
        af[mi].u[1] = *(const uint4*)&As[row * LDA + kb + 16];
      }
#pragma unroll
      for (int ni = 0; ni < 2; ++ni) {
        // B 32x16: lane = column N, lanes 0-15 K 0-7 & 16-23, lanes 16-31 K 8-15 & 24-31
        int nn = wn + ni * 16 + lrow;
        int kb = ks * 32 + hisel * 8;
        bf[ni].u[0] = *(const uint4*)&Bs[nn * LDB + kb];
        bf[ni].u[1] = *(const uint4*)&Bs[nn * LDB + kb + 16];
      }
#pragma unroll
      for (int mi = 0; mi < 2; ++mi)
#pragma unroll
        for (int ni = 0; ni < 2; ++ni)
          acc[mi][ni] = __builtin_amdgcn_wmma_f32_16x16x32_f16(
              false, af[mi].h, false, bf[ni].h, (short)0, acc[mi][ni], false,
              false);
    }

    // ---- ADC quantization: accumulate integer ADC codes (exact in f32) ----
#pragma unroll
    for (int mi = 0; mi < 2; ++mi)
#pragma unroll
      for (int ni = 0; ni < 2; ++ni)
#pragma unroll
        for (int e = 0; e < 8; ++e) {
          float v = acc[mi][ni][e] * inv_sps;
          v = fminf(fmaxf(v, -128.f), 127.f);
          tot[mi][ni][e] += rintf(v);  // s_ps folded into epilogue scale
        }
  }

  // ---- epilogue: dequantize per source matrix (s_ps * sx * sw), add bias ----
  const float sx = __uint_as_float(scales[0]) * (1.f / 127.f);
  const float swm[3] = {__uint_as_float(scales[1]) * (1.f / 127.f),
                        __uint_as_float(scales[2]) * (1.f / 127.f),
                        __uint_as_float(scales[3]) * (1.f / 127.f)};
#pragma unroll
  for (int mi = 0; mi < 2; ++mi)
#pragma unroll
    for (int ni = 0; ni < 2; ++ni) {
      int col = nblk * BN + wn + ni * 16 + lrow;
      int mat = col / COUT1;  // 0=q 1=k 2=v
      int cc  = col - mat * COUT1;
      const float* bias = (mat == 0) ? bq : (mat == 1) ? bk : bv;
      float sc = s_ps * sx * swm[mat];
      float bb = bias[cc];
#pragma unroll
      for (int e = 0; e < 8; ++e) {
        // C/D 16x16 f32 layout: VGPR e -> M = e + 8*hisel, N = lane&15
        int row = mblk * BM + wm + mi * 16 + e + hisel * 8;
        out[(size_t)row * NOUT + col] = tot[mi][ni][e] * sc + bb;
      }
    }
}

// ---------------- host launcher ----------------

extern "C" void kernel_launch(void* const* d_in, const int* in_sizes, int n_in,
                              void* d_out, int out_size, void* d_ws,
                              size_t ws_size, hipStream_t stream) {
  const float* x  = (const float*)d_in[0];
  const float* wq = (const float*)d_in[1];
  const float* wk = (const float*)d_in[2];
  const float* wv = (const float*)d_in[3];
  const float* bq = (const float*)d_in[4];
  const float* bk = (const float*)d_in[5];
  const float* bv = (const float*)d_in[6];
  const float* nq = (const float*)d_in[7];
  const float* nk = (const float*)d_in[8];
  const float* nv = (const float*)d_in[9];
  float* out = (float*)d_out;

  // workspace layout: [0,256) scale bits; then xq f16; then concat wq f16
  unsigned* scales = (unsigned*)d_ws;
  _Float16* xqbuf  = (_Float16*)((char*)d_ws + 256);
  _Float16* wqbuf  = xqbuf + (size_t)BT * CIN;

  k_zero_scales<<<1, 32, 0, stream>>>(scales);

  k_absmax<<<1024, 256, 0, stream>>>(x, BT * CIN, scales + 0);
  k_absmax<<<288, 256, 0, stream>>>(wq, COUT1 * CIN, scales + 1);
  k_absmax<<<288, 256, 0, stream>>>(wk, COUT1 * CIN, scales + 2);
  k_absmax<<<288, 256, 0, stream>>>(wv, COUT1 * CIN, scales + 3);

  k_quant_x<<<(BT * CIN / 4 + 255) / 256, 256, 0, stream>>>(x, scales, xqbuf);
  dim3 gw((COUT1 * CIN / 4 + 255) / 256, 3);
  k_quant_w<<<gw, 256, 0, stream>>>(wq, wk, wv, nq, nk, nv, scales, wqbuf);

  dim3 gg(NOUT / BN, BT / BM);  // 36 x 64
  k_cim_gemm<<<gg, 256, 0, stream>>>(xqbuf, wqbuf, scales, bq, bk, bv, out);
}
